// CTRL_Model_71691594105543
// MI455X (gfx1250) — compile-verified, matrix operand order checked by reading, not run on hardware
//
#include <hip/hip_runtime.h>

typedef __attribute__((ext_vector_type(16))) __bf16 bf16x16;
typedef __attribute__((ext_vector_type(8)))  __bf16 bf16x8;
typedef __attribute__((ext_vector_type(8)))  float  f32x8;

union ABu { bf16x8 h[2]; bf16x16 v; };

// ---------------------------------------------------------------------------
// Prep: build combined/padded weights in workspace.
//   Wgc[n,d] = W1[n, D+d] + W1[n, 2D+d]        (multiplies c[j])
//   Wgt[n,d] = W1[n, D+d] + W1[n, 3D+d]        (multiplies t[i])
//   W1a[n,d] = bf16(W1[n, d])                  (bilinear term), rows >=1000 zero
//   W2p[ch,k] = W2[ch,k] (k<1000) else 0       (3 x 1024)
// ---------------------------------------------------------------------------
__global__ __launch_bounds__(256) void prep_weights(
    const float* __restrict__ W1, const float* __restrict__ W2,
    float* __restrict__ Wgc, float* __restrict__ Wgt,
    __bf16* __restrict__ W1a, float* __restrict__ W2p)
{
    const int idx = blockIdx.x * 256 + threadIdx.x;   // 0 .. 1024*1024-1
    const int n = idx >> 10;
    const int d = idx & 1023;
    const bool valid = (n < 1000);
    const size_t row = (size_t)n * 4096;
    float a  = valid ? W1[row + d]        : 0.0f;
    float bb = valid ? W1[row + 1024 + d] : 0.0f;
    float cc = valid ? W1[row + 2048 + d] : 0.0f;
    float dd = valid ? W1[row + 3072 + d] : 0.0f;
    W1a[idx] = (__bf16)a;
    Wgc[idx] = bb + cc;
    Wgt[idx] = bb + dd;
    if (idx < 3072) {
        const int ch = idx >> 10, k = idx & 1023;
        W2p[idx] = (k < 1000) ? W2[ch * 1000 + k] : 0.0f;
    }
}

// ---------------------------------------------------------------------------
// Initialize out[i,j,ch] = b2[ch]; main kernel atomically accumulates on top.
// Re-run on every launch -> replay-deterministic.
// ---------------------------------------------------------------------------
__global__ __launch_bounds__(256) void init_out(
    float* __restrict__ out, const float* __restrict__ b2)
{
    const int idx = blockIdx.x * 256 + threadIdx.x;   // 0 .. 196607
    out[idx] = b2[idx % 3];
}

// ---------------------------------------------------------------------------
// FP32 tiled GEMM:  C[M,N] = A[M,K] * B[N,K]^T + bias   (M=256, N=1024 here)
// 64x64 tile / 256 threads / 4x4 microtile / K-step 16.  K must be mult of 16.
// ---------------------------------------------------------------------------
__global__ __launch_bounds__(256) void gemm_tn(
    const float* __restrict__ A, int lda,
    const float* __restrict__ B, int ldb,
    const float* __restrict__ bias, int bias_len,
    float* __restrict__ C, int ldc, int K)
{
    __shared__ float As[64][17];
    __shared__ float Bs[64][17];
    const int tid = threadIdx.x;
    const int tx = tid & 15, ty = tid >> 4;
    const int blockN = blockIdx.x * 64;
    const int blockM = blockIdx.y * 64;

    float acc[4][4];
    #pragma unroll
    for (int u = 0; u < 4; ++u)
        #pragma unroll
        for (int w = 0; w < 4; ++w) acc[u][w] = 0.0f;

    for (int kk = 0; kk < K; kk += 16) {
        #pragma unroll
        for (int e = 0; e < 4; ++e) {
            const int idx = tid + e * 256;
            const int r = idx >> 4, kloc = idx & 15;
            As[r][kloc] = A[(size_t)(blockM + r) * lda + kk + kloc];
            Bs[r][kloc] = B[(size_t)(blockN + r) * ldb + kk + kloc];
        }
        __syncthreads();
        #pragma unroll
        for (int k = 0; k < 16; ++k) {
            float ar[4], br[4];
            #pragma unroll
            for (int u = 0; u < 4; ++u) ar[u] = As[ty * 4 + u][k];
            #pragma unroll
            for (int w = 0; w < 4; ++w) br[w] = Bs[tx * 4 + w][k];
            #pragma unroll
            for (int u = 0; u < 4; ++u)
                #pragma unroll
                for (int w = 0; w < 4; ++w) acc[u][w] = fmaf(ar[u], br[w], acc[u][w]);
        }
        __syncthreads();
    }
    #pragma unroll
    for (int u = 0; u < 4; ++u) {
        #pragma unroll
        for (int w = 0; w < 4; ++w) {
            const int n = blockN + tx * 4 + w;
            const float bt = (bias && n < bias_len) ? bias[n] : 0.0f;
            C[(size_t)(blockM + ty * 4 + u) * ldc + n] = acc[u][w] + bt;
        }
    }
}

// ---------------------------------------------------------------------------
// Row-wise L2 normalize in place: X[256][1024], one block per row.
// ---------------------------------------------------------------------------
__global__ __launch_bounds__(256) void l2norm_rows(float* __restrict__ X)
{
    __shared__ float red[256];
    const int row = blockIdx.x, tid = threadIdx.x;
    float s = 0.0f;
    #pragma unroll
    for (int d = tid; d < 1024; d += 256) {
        const float v = X[(size_t)row * 1024 + d];
        s += v * v;
    }
    red[tid] = s;
    __syncthreads();
    for (int off = 128; off > 0; off >>= 1) {
        if (tid < off) red[tid] += red[tid + off];
        __syncthreads();
    }
    const float scale = 1.0f / fmaxf(sqrtf(red[0]), 1e-12f);
    #pragma unroll
    for (int d = tid; d < 1024; d += 256)
        X[(size_t)row * 1024 + d] *= scale;
}

// ---------------------------------------------------------------------------
// Fused main kernel (bf16 WMMA), 64-pair M-tile for 2x B reuse:
//   grid = (2, 1024): blockIdx.y -> 64 consecutive pairs (same i),
//                     blockIdx.x -> k_out half [x*512, x*512+512)
//   A tile p[m,d] = c[j,d]*t[i,d] (64x1024 bf16 = 128 KB LDS)
//   Each wave: M=64 (4 m-tiles) x N=64 (4 n-tiles), acc = 128 VGPRs;
//   each 32-byte B fragment feeds 4 WMMAs.
//   Epilogue: relu + W2 contraction, LDS reduce, global atomic into out(=b2).
// ---------------------------------------------------------------------------
__global__ __launch_bounds__(256) void fused_bilinear_wmma(
    const float* __restrict__ c, const float* __restrict__ t,
    const __bf16* __restrict__ W1a,
    const float* __restrict__ Gc, const float* __restrict__ Gt,
    const float* __restrict__ W2p, float* __restrict__ out)
{
    __shared__ __align__(16) __bf16 Atile[64 * 1024];   // 128 KB

    const int tid = threadIdx.x;
    const int wgBase = blockIdx.y * 64;      // first flattened pair index
    const int i = wgBase >> 8;               // sentence row (same for all 64)
    const int jbase = wgBase & 255;          // clip rows jbase..jbase+63

    // ---- build A tile: p[m][d] = c[jbase+m][d] * t[i][d] (fp32 mul -> bf16)
    {
        const int m = tid >> 2;
        const int dbase = (tid & 3) * 256;
        const float* __restrict__ crow = c + (size_t)(jbase + m) * 1024 + dbase;
        const float* __restrict__ trow = t + (size_t)i * 1024 + dbase;
        __bf16* arow = Atile + m * 1024 + dbase;
        #pragma unroll 8
        for (int d = 0; d < 256; ++d)
            arow[d] = (__bf16)(crow[d] * trow[d]);
    }
    __syncthreads();

    const int wave  = tid >> 5;
    const int lane  = tid & 31;
    const int lhalf = lane >> 4;     // 0: lanes 0-15, 1: lanes 16-31
    const int l16   = lane & 15;
    const int nBase = blockIdx.x * 512 + wave * 64;   // wave's k_out slice

    const f32x8 zacc = {0.f, 0.f, 0.f, 0.f, 0.f, 0.f, 0.f, 0.f};
    f32x8 acc[4][4];
    #pragma unroll
    for (int mt = 0; mt < 4; ++mt)
        #pragma unroll
        for (int nt = 0; nt < 4; ++nt) acc[mt][nt] = zacc;

    // ---- K loop: 1024 / 32 steps, 16 WMMAs each
    for (int kk = 0; kk < 1024; kk += 32) {
        // 16-bit A layout: lanes 0-15 hold K chunks [kk, kk+8) and [kk+16, kk+24); lanes 16-31 +8
        const int c0 = kk + lhalf * 8;
        const int c1 = kk + 16 + lhalf * 8;
        ABu a[4];
        #pragma unroll
        for (int mt = 0; mt < 4; ++mt) {
            const __bf16* arow = Atile + (size_t)(mt * 16 + l16) * 1024;
            a[mt].h[0] = *(const bf16x8*)(arow + c0);
            a[mt].h[1] = *(const bf16x8*)(arow + c1);
        }
        #pragma unroll
        for (int nt = 0; nt < 4; ++nt) {
            // B[k, n] = W1a[n, k]; lane holds column n = nBase + nt*16 + l16
            const __bf16* __restrict__ brow =
                W1a + (size_t)(nBase + nt * 16 + l16) * 1024;
            ABu bf;
            bf.h[0] = *(const bf16x8*)(brow + c0);
            bf.h[1] = *(const bf16x8*)(brow + c1);
            // keep this lane's B row streaming ahead in L2 (global_prefetch_b8)
            __builtin_prefetch((const void*)(brow + c0 + 256), 0, 1);
            #pragma unroll
            for (int mt = 0; mt < 4; ++mt)
                acc[mt][nt] = __builtin_amdgcn_wmma_f32_16x16x32_bf16(
                    false, a[mt].v, false, bf.v, (short)0, acc[mt][nt],
                    false, false);
        }
    }

    // ---- fused epilogue: relu + 3-channel W2 contraction, reduce in LDS
    __syncthreads();
    float* outAcc = reinterpret_cast<float*>(Atile);  // reuse LDS: 64*3 floats
    if (tid < 192) outAcc[tid] = 0.0f;
    __syncthreads();

    #pragma unroll
    for (int mt = 0; mt < 4; ++mt) {
        float s0[8], s1[8], s2[8];
        #pragma unroll
        for (int v = 0; v < 8; ++v) { s0[v] = 0.f; s1[v] = 0.f; s2[v] = 0.f; }
        #pragma unroll
        for (int nt = 0; nt < 4; ++nt) {
            const int kg = nBase + nt * 16 + l16;   // lane's N within tile
            const float gt = Gt[(size_t)i * 1024 + kg];
            const float w0 = W2p[kg];
            const float w1 = W2p[1024 + kg];
            const float w2 = W2p[2048 + kg];
            #pragma unroll
            for (int v = 0; v < 8; ++v) {
                // C layout: VGPR v -> M = v (+8 for lanes 16-31)
                const int m = mt * 16 + v + (lhalf << 3);
                float x = acc[mt][nt][v] + gt +
                          Gc[(size_t)(jbase + m) * 1024 + kg];
                x = fmaxf(x, 0.0f);
                s0[v] = fmaf(w0, x, s0[v]);
                s1[v] = fmaf(w1, x, s1[v]);
                s2[v] = fmaf(w2, x, s2[v]);
            }
        }
        #pragma unroll
        for (int v = 0; v < 8; ++v) {
            const int m = mt * 16 + v + (lhalf << 3);
            atomicAdd(&outAcc[m * 3 + 0], s0[v]);
            atomicAdd(&outAcc[m * 3 + 1], s1[v]);
            atomicAdd(&outAcc[m * 3 + 2], s2[v]);
        }
    }
    __syncthreads();
    // combine the two k_out halves (blockIdx.x = 0/1) in global memory
    if (tid < 192) {
        const int m = tid / 3, ch = tid % 3;
        atomicAdd(&out[(size_t)(wgBase + m) * 3 + ch], outAcc[tid]);
    }
}

// ---------------------------------------------------------------------------
extern "C" void kernel_launch(void* const* d_in, const int* in_sizes, int n_in,
                              void* d_out, int out_size, void* d_ws, size_t ws_size,
                              hipStream_t stream)
{
    (void)in_sizes; (void)n_in; (void)out_size; (void)ws_size;
    const float* visual   = (const float*)d_in[0];
    const float* sentence = (const float*)d_in[1];
    const float* Wv = (const float*)d_in[2];
    const float* bv = (const float*)d_in[3];
    const float* Ws = (const float*)d_in[4];
    const float* bs = (const float*)d_in[5];
    const float* W1 = (const float*)d_in[6];
    const float* b1 = (const float*)d_in[7];
    const float* W2 = (const float*)d_in[8];
    const float* b2 = (const float*)d_in[9];
    float* out = (float*)d_out;

    char* base = (char*)d_ws;
    float*  c_buf = (float*)(base);                          // 1 MiB
    float*  t_buf = (float*)(base + (1ull  << 20));          // 1 MiB
    float*  Wgc   = (float*)(base + (2ull  << 20));          // 4 MiB
    float*  Wgt   = (float*)(base + (6ull  << 20));          // 4 MiB
    __bf16* W1a   = (__bf16*)(base + (10ull << 20));         // 2 MiB
    float*  Gc    = (float*)(base + (12ull << 20));          // 1 MiB
    float*  Gt    = (float*)(base + (13ull << 20));          // 1 MiB
    float*  W2p   = (float*)(base + (14ull << 20));          // 12 KiB

    prep_weights<<<4096, 256, 0, stream>>>(W1, W2, Wgc, Wgt, W1a, W2p);
    init_out<<<768, 256, 0, stream>>>(out, b2);              // out = b2

    dim3 g(16, 4);  // N=1024 / 64, M=256 / 64
    gemm_tn<<<g, 256, 0, stream>>>(visual, 12288, Wv, 12288, bv, 1024,
                                   c_buf, 1024, 12288);
    gemm_tn<<<g, 256, 0, stream>>>(sentence, 4800, Ws, 4800, bs, 1024,
                                   t_buf, 1024, 4800);
    l2norm_rows<<<256, 256, 0, stream>>>(c_buf);
    l2norm_rows<<<256, 256, 0, stream>>>(t_buf);
    gemm_tn<<<g, 256, 0, stream>>>(c_buf, 1024, Wgc, 1024, nullptr, 0,
                                   Gc, 1024, 1024);
    gemm_tn<<<g, 256, 0, stream>>>(t_buf, 1024, Wgt, 1024, b1, 1000,
                                   Gt, 1024, 1024);

    fused_bilinear_wmma<<<dim3(2, 1024), 256, 0, stream>>>(
        c_buf, t_buf, W1a, Gc, Gt, W2p, out);
}
